// TransformerConvLayer_38173669327085
// MI455X (gfx1250) — matrix operand order repaired and checked by reading, not compiled, not run on hardware
//
#include <hip/hip_runtime.h>
#include <hip/hip_bf16.h>
#include <math.h>

// Problem sizes (fixed by the reference).
#define NN  50000    // nodes
#define NE  400000   // edges
#define DD  128      // node dim
#define HH  8        // heads
#define CC  16       // channels per head
#define HCC 128      // H*C
#define EDM 64       // edge dim

typedef __attribute__((ext_vector_type(16))) __bf16        bf16x16;
typedef __attribute__((ext_vector_type(2)))  __bf16        bf16x2;
typedef __attribute__((ext_vector_type(8)))  float         floatx8;
typedef __attribute__((ext_vector_type(4)))  float         float4v;
typedef __attribute__((ext_vector_type(2)))  unsigned int  uint2v;
typedef __attribute__((ext_vector_type(8)))  unsigned int  uint8v;

__device__ __forceinline__ float gelu_f(float x) {
  return 0.5f * x * (1.0f + erff(x * 0.70710678118654752f));
}

// Two f32 -> one packed bf16 pair; bf16x2 bit-cast pattern-matches to a
// single v_cvt_pk_bf16_f32.
__device__ __forceinline__ unsigned int pack_bf16(float a, float b) {
  bf16x2 p;
  p[0] = (__bf16)a;
  p[1] = (__bf16)b;
  return __builtin_bit_cast(unsigned int, p);
}

// segment_max via integer-compare trick (works for mixed-sign floats, init = -inf).
__device__ __forceinline__ void atomicMaxFloat(float* addr, float val) {
  if (val >= 0.0f) atomicMax((int*)addr, __float_as_int(val));
  else             atomicMin((unsigned int*)addr, (unsigned int)__float_as_int(val));
}

// ---------------------------------------------------------------------------
// GEMM: C[M,Nc] = act(A[M,K] @ B[K,Nc] + bias), fp32 in/out, bf16 WMMA inner
// product (v_wmma_f32_16x16x32_bf16), f32 accumulation.
//
// Requirements (hold for every call in this layer): M%16==0, K%64==0, Nc%BN==0.
// Block = BN/16 waves; block tile 16 x BN; each wave owns one 16x16 C tile;
// K loop steps by 64 -> two WMMAs per barrier pair. Staging task counts are
// exact multiples of the thread count for both BN=128 (256 thr) and BN=64
// (128 thr), so all staging loops fully unroll with no EXEC masking.
//
// LDS: pair-packed bf16 as uint, row pitch 34 uints (8B-aligned, 34*l mod 64
// distinct for l=0..15 -> conflict-free fragment loads).
//   Asp[r][p] = K-pair p of A row r          (16 rows x 32 pairs)
//   Bsp[n][p] = K-pair p of B column n0+n    (transposed: BN cols x 32 pairs)
// Fragment packing per CDNA5 ISA (16-bit A 16x32, B 32x16):
//   A lane(l,half), k-tile t: pairs Asp[l][16t + half*4 + 0..3]
//                               and Asp[l][16t + 8 + half*4 + 0..3]
//   B lane(l,half), k-tile t: pairs Bsp[col][16t + half*4 + 0..3]
//                               and Bsp[col][16t + 8 + half*4 + 0..3]
//   C vgpr r -> (row = r + 8*half, col = lane%16)
// ---------------------------------------------------------------------------
template<int BN>
__global__ __launch_bounds__(2 * BN) void gemm_bf16_wmma(
    const float* __restrict__ A, const float* __restrict__ B,
    const float* __restrict__ bias, float* __restrict__ C,
    int M, int K, int Nc, int act)
{
  constexpr int NTHR = 2 * BN;          // 32 * (BN/16) waves
  __shared__ unsigned int Asp[16][34];
  __shared__ unsigned int Bsp[BN][34];

  const int tid  = threadIdx.x;
  const int wave = tid >> 5;
  const int lane = tid & 31;
  const int l    = lane & 15;
  const int half = lane >> 4;
  const int m0   = blockIdx.x * 16;
  const int n0   = blockIdx.y * BN;

  floatx8 acc = {};

  for (int k0 = 0; k0 < K; k0 += 64) {
    // ---- stage A tile: 16x64 f32 = 256 float4 tasks (exact multiple) ----
    #pragma unroll
    for (int t = tid; t < 256; t += NTHR) {
      int r = t >> 4, q = t & 15;
      const float4v v = *(const float4v*)(A + (size_t)(m0 + r) * K + k0 + q * 4);
      uint2v p; p[0] = pack_bf16(v[0], v[1]); p[1] = pack_bf16(v[2], v[3]);
      *(uint2v*)&Asp[r][q * 2] = p;     // ds_store_b64
    }
    // ---- stage B tile transposed: 32 k-pairs x BN/4 col groups ----
    #pragma unroll
    for (int t = tid; t < 32 * (BN / 4); t += NTHR) {
      int kp = t / (BN / 4), nq = t % (BN / 4);
      const float4v r0 = *(const float4v*)(B + (size_t)(k0 + 2 * kp)     * Nc + n0 + nq * 4);
      const float4v r1 = *(const float4v*)(B + (size_t)(k0 + 2 * kp + 1) * Nc + n0 + nq * 4);
      #pragma unroll
      for (int j = 0; j < 4; ++j)
        Bsp[nq * 4 + j][kp] = pack_bf16(r0[j], r1[j]);
    }
    // Prefetch next K tile (global_prefetch_b8).
    if (k0 + 64 < K) {
      __builtin_prefetch(A + (size_t)(m0 + (tid & 15)) * K + k0 + 64, 0, 1);
      __builtin_prefetch(B + (size_t)(k0 + 64 + (tid >> 3)) * Nc + n0, 0, 1);
    }
    __syncthreads();

    const unsigned int* ar = &Asp[l][0];
    const unsigned int* br = &Bsp[wave * 16 + l][0];

    // ---- two K=32 WMMA steps per staged tile ----
    #pragma unroll
    for (int t = 0; t < 2; ++t) {
      uint8v au, bu;
      {
        const unsigned int* p = ar + 16 * t;
        uint2v a0 = *(const uint2v*)(p + half * 4);
        uint2v a1 = *(const uint2v*)(p + half * 4 + 2);
        uint2v a2 = *(const uint2v*)(p + 8 + half * 4);
        uint2v a3 = *(const uint2v*)(p + 8 + half * 4 + 2);
        au[0] = a0[0]; au[1] = a0[1]; au[2] = a1[0]; au[3] = a1[1];
        au[4] = a2[0]; au[5] = a2[1]; au[6] = a3[0]; au[7] = a3[1];
      }
      {
        const unsigned int* p = br + 16 * t;
        uint2v b0 = *(const uint2v*)(p + half * 4);
        uint2v b1 = *(const uint2v*)(p + half * 4 + 2);
        uint2v b2 = *(const uint2v*)(p + 8 + half * 4);
        uint2v b3 = *(const uint2v*)(p + 8 + half * 4 + 2);
        bu[0] = b0[0]; bu[1] = b0[1]; bu[2] = b1[0]; bu[3] = b1[1];
        bu[4] = b2[0]; bu[5] = b2[1]; bu[6] = b3[0]; bu[7] = b3[1];
      }
      bf16x16 af = __builtin_bit_cast(bf16x16, au);
      bf16x16 bf = __builtin_bit_cast(bf16x16, bu);
      acc = __builtin_amdgcn_wmma_f32_16x16x32_bf16(
          /*neg_a=*/false, af, /*neg_b=*/false, bf,
          /*c_mod=*/(short)0, acc, /*reuse_a=*/false, /*reuse_b=*/false);
    }
    __syncthreads();
  }

  // ---- epilogue: bias + optional GELU, coalesced row stores ----
  const int col = n0 + wave * 16 + l;
  const float bv = bias ? bias[col] : 0.0f;
  #pragma unroll
  for (int r = 0; r < 8; ++r) {
    int row = m0 + half * 8 + r;
    float v = acc[r] + bv;
    if (act == 1) v = gelu_f(v);
    C[(size_t)row * Nc + col] = v;
  }
}

// ---------------------------------------------------------------------------
// Attention pass A: Knew = K[dst]Wk1 + K[src]Wk2 + e Wk3 + bke ; logits ;
// segment_max(dst) via float atomic max.
// ---------------------------------------------------------------------------
__global__ __launch_bounds__(256) void attn_logits_kernel(
    const float* __restrict__ Q, const float* __restrict__ K,
    const float* __restrict__ e, const int* __restrict__ ei,
    const float* __restrict__ Wke, const float* __restrict__ bke,
    float* __restrict__ logits, float* __restrict__ mx)
{
  __shared__ float Wk_s[3 * CC * CC];
  __shared__ float bk_s[CC];
  for (int i = threadIdx.x; i < 3 * CC * CC; i += blockDim.x) Wk_s[i] = Wke[i];
  if (threadIdx.x < CC) bk_s[threadIdx.x] = bke[threadIdx.x];
  __syncthreads();

  int idx = blockIdx.x * blockDim.x + threadIdx.x;
  if (idx >= NE * HH) return;
  int eid = idx / HH, h = idx % HH;
  int src = ei[eid];
  int dst = ei[NE + eid];

  const float* Kd = K + (size_t)dst * HCC + h * CC;
  const float* Ks = K + (size_t)src * HCC + h * CC;
  const float* Ev = e + (size_t)eid * HCC + h * CC;
  const float* Qi = Q + (size_t)dst * HCC + h * CC;

  float kd[CC], ks[CC], ev[CC], qv[CC];
  #pragma unroll
  for (int c = 0; c < CC; ++c) { kd[c]=Kd[c]; ks[c]=Ks[c]; ev[c]=Ev[c]; qv[c]=Qi[c]; }

  float logit = 0.0f;
  #pragma unroll 4
  for (int c = 0; c < CC; ++c) {
    float kn = bk_s[c];
    for (int d = 0; d < CC; ++d)
      kn += kd[d] * Wk_s[d * CC + c]
          + ks[d] * Wk_s[(CC + d) * CC + c]
          + ev[d] * Wk_s[(2 * CC + d) * CC + c];
    logit += qv[c] * kn;
  }
  logit *= 0.25f; // 1/sqrt(C), C=16
  logits[idx] = logit;
  atomicMaxFloat(&mx[dst * HH + h], logit);
}

// Pass B: exp(logit - mx[dst]) and segment_sum(dst).
__global__ __launch_bounds__(256) void attn_softmax_kernel(
    float* __restrict__ logits, const float* __restrict__ mx,
    float* __restrict__ den, const int* __restrict__ ei)
{
  int idx = blockIdx.x * blockDim.x + threadIdx.x;
  if (idx >= NE * HH) return;
  int eid = idx / HH, h = idx % HH;
  int dst = ei[NE + eid];
  float ex = __expf(logits[idx] - mx[dst * HH + h]);
  logits[idx] = ex;
  atomicAdd(&den[dst * HH + h], ex);
}

// Pass C: alpha; Vnew; scatter-add msg into out_agg; edge_store = e*alpha (in place).
__global__ __launch_bounds__(256) void attn_aggregate_kernel(
    const float* __restrict__ V, float* e, const float* __restrict__ ex,
    const float* __restrict__ den, const int* __restrict__ ei,
    const float* __restrict__ Wve, const float* __restrict__ bve,
    float* __restrict__ out_agg)
{
  __shared__ float Wv_s[3 * CC * CC];
  __shared__ float bv_s[CC];
  for (int i = threadIdx.x; i < 3 * CC * CC; i += blockDim.x) Wv_s[i] = Wve[i];
  if (threadIdx.x < CC) bv_s[threadIdx.x] = bve[threadIdx.x];
  __syncthreads();

  int idx = blockIdx.x * blockDim.x + threadIdx.x;
  if (idx >= NE * HH) return;
  int eid = idx / HH, h = idx % HH;
  int src = ei[eid];
  int dst = ei[NE + eid];

  float alpha = ex[idx] / (den[dst * HH + h] + 1e-16f);

  const float* Vd = V + (size_t)dst * HCC + h * CC;
  const float* Vs = V + (size_t)src * HCC + h * CC;
  float* Ev = e + (size_t)eid * HCC + h * CC;

  float vd[CC], vs[CC], ev[CC];
  #pragma unroll
  for (int c = 0; c < CC; ++c) { vd[c]=Vd[c]; vs[c]=Vs[c]; ev[c]=Ev[c]; }

  #pragma unroll 4
  for (int c = 0; c < CC; ++c) {
    float vn = bv_s[c];
    for (int d = 0; d < CC; ++d)
      vn += vd[d] * Wv_s[d * CC + c]
          + vs[d] * Wv_s[(CC + d) * CC + c]
          + ev[d] * Wv_s[(2 * CC + d) * CC + c];
    atomicAdd(&out_agg[(size_t)dst * HCC + h * CC + c], vn * alpha);
  }
  #pragma unroll
  for (int c = 0; c < CC; ++c) Ev[c] = ev[c] * alpha;  // edge_store
}

// Row LayerNorm + residual, one wave32 per row (W = 64 or 128).
__global__ __launch_bounds__(256) void ln_residual_kernel(
    const float* __restrict__ in, const float* __restrict__ res,
    const float* __restrict__ g, const float* __restrict__ b,
    float* __restrict__ out, int rows, int W)
{
  int wavesPerBlock = blockDim.x >> 5;
  int row  = blockIdx.x * wavesPerBlock + (threadIdx.x >> 5);
  int lane = threadIdx.x & 31;
  if (row >= rows) return;

  const float* r = in + (size_t)row * W;
  int per = W >> 5;        // 2 or 4
  float vals[4];
  float s = 0.0f, s2 = 0.0f;
  for (int i = 0; i < per; ++i) {
    float v = r[lane + i * 32];
    vals[i] = v; s += v; s2 += v * v;
  }
  for (int o = 16; o > 0; o >>= 1) {
    s  += __shfl_xor(s,  o, 32);
    s2 += __shfl_xor(s2, o, 32);
  }
  float m   = s / (float)W;
  float var = s2 / (float)W - m * m;
  float inv = rsqrtf(var + 1e-5f);
  for (int i = 0; i < per; ++i) {
    int c = lane + i * 32;
    out[(size_t)row * W + c] = (vals[i] - m) * inv * g[c] + b[c]
                             + res[(size_t)row * W + c];
  }
}

// BatchNorm column stats: stats[0..W)=sum, stats[W..2W)=sumsq (grid-stride + LDS atomics).
__global__ __launch_bounds__(256) void bn_stats_kernel(
    const float* __restrict__ y, int rows, int W, float* __restrict__ stats)
{
  __shared__ float ssum[128];
  __shared__ float ssq[128];
  for (int c = threadIdx.x; c < W; c += blockDim.x) { ssum[c] = 0.f; ssq[c] = 0.f; }
  __syncthreads();
  size_t total = (size_t)rows * W;
  for (size_t i = (size_t)blockIdx.x * blockDim.x + threadIdx.x; i < total;
       i += (size_t)gridDim.x * blockDim.x) {
    float v = y[i];
    int c = (int)(i % (size_t)W);
    atomicAdd(&ssum[c], v);
    atomicAdd(&ssq[c], v * v);
  }
  __syncthreads();
  for (int c = threadIdx.x; c < W; c += blockDim.x) {
    atomicAdd(&stats[c],     ssum[c]);
    atomicAdd(&stats[W + c], ssq[c]);
  }
}

// out = softplus(batchnorm(y)*g + b + res)
__global__ __launch_bounds__(256) void bn_apply_softplus_kernel(
    const float* __restrict__ y, const float* __restrict__ res,
    const float* __restrict__ stats, const float* __restrict__ g,
    const float* __restrict__ b, float* __restrict__ out, int rows, int W)
{
  size_t total = (size_t)rows * W;
  for (size_t i = (size_t)blockIdx.x * blockDim.x + threadIdx.x; i < total;
       i += (size_t)gridDim.x * blockDim.x) {
    int c = (int)(i % (size_t)W);
    float m   = stats[c] / (float)rows;
    float var = stats[W + c] / (float)rows - m * m;
    float v = (y[i] - m) * rsqrtf(var + 1e-5f) * g[c] + b[c] + res[i];
    out[i] = (v > 20.0f) ? v : log1pf(__expf(v));
  }
}

__global__ void init_neg_inf_kernel(float* p, int n) {
  int i = blockIdx.x * blockDim.x + threadIdx.x;
  if (i < n) p[i] = -__builtin_inff();
}

// ---------------------------------------------------------------------------
extern "C" void kernel_launch(void* const* d_in, const int* in_sizes, int n_in,
                              void* d_out, int out_size, void* d_ws, size_t ws_size,
                              hipStream_t stream) {
  (void)in_sizes; (void)n_in; (void)out_size; (void)ws_size;

  const float* x         = (const float*)d_in[0];
  const int*   ei        = (const int*)  d_in[1];
  const float* edge_attr = (const float*)d_in[2];
  const float* Wq  = (const float*)d_in[3];  const float* bq  = (const float*)d_in[4];
  const float* Wk  = (const float*)d_in[5];  const float* bk  = (const float*)d_in[6];
  const float* Wv  = (const float*)d_in[7];  const float* bv  = (const float*)d_in[8];
  const float* We  = (const float*)d_in[9];
  const float* Wke = (const float*)d_in[10]; const float* bke = (const float*)d_in[11];
  const float* Wve = (const float*)d_in[12]; const float* bve = (const float*)d_in[13];
  const float* Wfce= (const float*)d_in[14]; const float* bfce= (const float*)d_in[15];
  const float* lne_g=(const float*)d_in[16]; const float* lne_b=(const float*)d_in[17];
  const float* Wfcn= (const float*)d_in[18]; const float* bfcn= (const float*)d_in[19];
  const float* lnn_g=(const float*)d_in[20]; const float* lnn_b=(const float*)d_in[21];
  const float* W1n = (const float*)d_in[22]; const float* b1n = (const float*)d_in[23];
  const float* W2n = (const float*)d_in[24]; const float* b2n = (const float*)d_in[25];
  const float* W1e = (const float*)d_in[26]; const float* b1e = (const float*)d_in[27];
  const float* W2e = (const float*)d_in[28]; const float* b2e = (const float*)d_in[29];
  const float* bnn_g=(const float*)d_in[30]; const float* bnn_b=(const float*)d_in[31];
  const float* bne_g=(const float*)d_in[32]; const float* bne_b=(const float*)d_in[33];
  float* out = (float*)d_out;                // [N*D] x2 then [E*ED] e2

  // Workspace carve-out (~940 MB of fp32).
  float* wsf = (float*)d_ws;
  size_t o = 0;
  float* Qb      = wsf + o; o += (size_t)NN * HCC;   // reused as tmp_n after attn
  float* Kb      = wsf + o; o += (size_t)NN * HCC;   // reused as x_out
  float* Vb      = wsf + o; o += (size_t)NN * HCC;   // reused as y_n
  float* eb      = wsf + o; o += (size_t)NE * HCC;   // e, then edge_store (in place)
  float* logitsb = wsf + o; o += (size_t)NE * HH;    // logits, then ex (in place)
  float* mxb     = wsf + o; o += (size_t)NN * HH;
  float* denb    = wsf + o; o += (size_t)NN * HH;
  float* outagg  = wsf + o; o += (size_t)NN * HCC;
  float* tmpe    = wsf + o; o += (size_t)NE * EDM;   // fc-edge out, then y_e
  float* eoutb   = wsf + o; o += (size_t)NE * EDM;
  float* hbuf    = wsf + o; o += (size_t)NE * 4 * EDM; // FFN hidden (covers node case too)
  float* stats   = wsf + o; o += 256;

  auto gemm = [&](const float* A, const float* B, const float* bias, float* C,
                  int M, int K, int Nc, int act) {
    if (Nc % 128 == 0) {
      dim3 g((unsigned)(M / 16), (unsigned)(Nc / 128));
      gemm_bf16_wmma<128><<<g, 256, 0, stream>>>(A, B, bias, C, M, K, Nc, act);
    } else { // Nc % 64 == 0
      dim3 g((unsigned)(M / 16), (unsigned)(Nc / 64));
      gemm_bf16_wmma<64><<<g, 128, 0, stream>>>(A, B, bias, C, M, K, Nc, act);
    }
  };

  // ---- init segment buffers ----
  hipMemsetAsync(denb,   0, (size_t)NN * HH  * sizeof(float), stream);
  hipMemsetAsync(outagg, 0, (size_t)NN * HCC * sizeof(float), stream);
  init_neg_inf_kernel<<<(NN * HH + 255) / 256, 256, 0, stream>>>(mxb, NN * HH);

  // ---- projections (WMMA GEMMs) ----
  gemm(x,         Wq, bq,      Qb, NN, DD,  HCC, 0);
  gemm(x,         Wk, bk,      Kb, NN, DD,  HCC, 0);
  gemm(x,         Wv, bv,      Vb, NN, DD,  HCC, 0);
  gemm(edge_attr, We, nullptr, eb, NE, EDM, HCC, 0);

  // ---- attention (segment softmax over dst) ----
  const int EH_BLOCKS = (NE * HH + 255) / 256;
  attn_logits_kernel   <<<EH_BLOCKS, 256, 0, stream>>>(Qb, Kb, eb, ei, Wke, bke, logitsb, mxb);
  attn_softmax_kernel  <<<EH_BLOCKS, 256, 0, stream>>>(logitsb, mxb, denb, ei);
  attn_aggregate_kernel<<<EH_BLOCKS, 256, 0, stream>>>(Vb, eb, logitsb, denb, ei, Wve, bve, outagg);

  // ---- node path: FC + LN + residual, FFN, BN + residual, softplus ----
  gemm(outagg, Wfcn, bfcn, Qb, NN, HCC, DD, 0);                   // tmp_n
  ln_residual_kernel<<<(NN + 7) / 8, 256, 0, stream>>>(Qb, x, lnn_g, lnn_b, Kb, NN, DD); // x_out
  gemm(Kb,   W1n, b1n, hbuf, NN, DD,     4 * DD, 1);              // gelu hidden
  gemm(hbuf, W2n, b2n, Vb,   NN, 4 * DD, DD,     0);              // y_n
  hipMemsetAsync(stats, 0, 256 * sizeof(float), stream);
  bn_stats_kernel        <<<2048, 256, 0, stream>>>(Vb, NN, DD, stats);
  bn_apply_softplus_kernel<<<2048, 256, 0, stream>>>(Vb, Kb, stats, bnn_g, bnn_b, out, NN, DD);

  // ---- edge path ----
  gemm(eb, Wfce, bfce, tmpe, NE, HCC, EDM, 0);                    // fc on edge_store
  ln_residual_kernel<<<(NE + 7) / 8, 256, 0, stream>>>(tmpe, edge_attr, lne_g, lne_b, eoutb, NE, EDM);
  gemm(eoutb, W1e, b1e, hbuf, NE, EDM,     4 * EDM, 1);           // gelu hidden
  gemm(hbuf,  W2e, b2e, tmpe, NE, 4 * EDM, EDM,     0);           // y_e
  hipMemsetAsync(stats, 0, 256 * sizeof(float), stream);
  bn_stats_kernel        <<<2048, 256, 0, stream>>>(tmpe, NE, EDM, stats);
  bn_apply_softplus_kernel<<<2048, 256, 0, stream>>>(tmpe, eoutb, stats, bne_g, bne_b,
                                                     out + (size_t)NN * DD, NE, EDM);
}
